// QuantLinearSim_18880676233635
// MI455X (gfx1250) — compile-verified, matrix operand order checked by reading, not run on hardware
//
#include <hip/hip_runtime.h>

typedef _Float16 v16h __attribute__((ext_vector_type(16)));
typedef _Float16 v8h  __attribute__((ext_vector_type(8)));
typedef _Float16 v4h  __attribute__((ext_vector_type(4)));
typedef float    v8f  __attribute__((ext_vector_type(8)));

static constexpr int ND   = 4096;   // square problem: tokens = in = out = 4096
static constexpr int MAXO = 64;     // max stored outliers per row (actual ~6)

// ---------------------------------------------------------------------------
// CDNA5 async global->LDS helpers (ASYNCcnt-tracked, bypasses VGPRs)
// ---------------------------------------------------------------------------
__device__ __forceinline__ uint32_t lds_off_u32(const void* p) {
  // LDS-aperture flat addresses carry the wave-relative LDS offset in [31:0]
  return (uint32_t)(uintptr_t)p;
}

__device__ __forceinline__ void async_copy_b256(uint32_t lds, uint64_t gaddr) {
  // 32B per lane: two b128 async copies; offset:16 applies to BOTH LDS and
  // global sides (ISA 15.18.3).
  asm volatile(
      "global_load_async_to_lds_b128 %0, %1, off\n\t"
      "global_load_async_to_lds_b128 %0, %1, off offset:16"
      :: "v"(lds), "v"(gaddr) : "memory");
}

__device__ __forceinline__ void wait_async0() {
  asm volatile("s_wait_asynccnt 0x0" ::: "memory");
}

// ---------------------------------------------------------------------------
// Kernel 1: per-row NF4 quantization (sort -> quantiles -> scale -> snap)
// One 256-thread block per weight row.
// ---------------------------------------------------------------------------
__global__ __launch_bounds__(256) void nf4_quant_kernel(
    const float* __restrict__ W, const float* __restrict__ nf,
    _Float16* __restrict__ Qh, float* __restrict__ rangev, float* __restrict__ offsetv,
    int* __restrict__ ocount, int* __restrict__ oidx, float* __restrict__ oval)
{
  __shared__ float s_orig[ND];
  __shared__ float s_sort[ND];
  __shared__ float s_red [256];
  __shared__ float s_red2[256];
  __shared__ int   s_scan[256];
  __shared__ float s_b[8];
  __shared__ float s_nf[16];

  const int row = blockIdx.x;
  const int tid = threadIdx.x;
  const float* wrow = W + (size_t)row * ND;

  if (tid < 16) s_nf[tid] = nf[tid];
  for (int i = tid; i < ND / 4; i += 256) {
    float4 v = ((const float4*)wrow)[i];
    ((float4*)s_orig)[i] = v;
    ((float4*)s_sort)[i] = v;
  }
  __syncthreads();

  // Bitonic sort (ascending), 4096 elements, 256 threads.
  for (int k = 2; k <= ND; k <<= 1) {
    for (int j = k >> 1; j > 0; j >>= 1) {
      for (int i = tid; i < ND; i += 256) {
        int ixj = i ^ j;
        if (ixj > i) {
          float a = s_sort[i], b = s_sort[ixj];
          bool asc = ((i & k) == 0);
          if ((asc && a > b) || (!asc && a < b)) { s_sort[i] = b; s_sort[ixj] = a; }
        }
      }
      __syncthreads();
    }
  }

  if (tid == 0) {
    // jnp.quantile linear interp: h = t*(n-1); t=0.0005 -> 2.0475, t=0.9995 -> 4092.9525
    float lower = s_sort[2]    + 0.0475f * (s_sort[3]    - s_sort[2]);
    float upper = s_sort[4092] + 0.9525f * (s_sort[4093] - s_sort[4092]);
    float med   = 0.5f * (s_sort[2047] + s_sort[2048]);
    s_b[0] = lower; s_b[1] = upper; s_b[2] = med;
  }
  __syncthreads();
  const float lower = s_b[0], upper = s_b[1], med = s_b[2];

  // min/max of outlier-replaced row + per-thread outlier count (contiguous chunks
  // keep the compaction order == k order => deterministic).
  float mx = -3.4e38f, mn = 3.4e38f;
  int cnt = 0;
  const int base = tid * 16;
  for (int i = 0; i < 16; ++i) {
    float v = s_orig[base + i];
    bool m = (v <= lower) || (v >= upper);
    cnt += m ? 1 : 0;
    float t = m ? med : v;
    mx = fmaxf(mx, t); mn = fminf(mn, t);
  }
  s_red[tid] = mx; s_red2[tid] = mn; s_scan[tid] = cnt;
  __syncthreads();
  for (int s = 128; s > 0; s >>= 1) {
    if (tid < s) {
      s_red [tid] = fmaxf(s_red [tid], s_red [tid + s]);
      s_red2[tid] = fminf(s_red2[tid], s_red2[tid + s]);
    }
    __syncthreads();
  }
  // Hillis-Steele inclusive scan of outlier counts (deterministic compaction).
  for (int off = 1; off < 256; off <<= 1) {
    int v   = s_scan[tid];
    int add = (tid >= off) ? s_scan[tid - off] : 0;
    __syncthreads();
    s_scan[tid] = v + add;
    __syncthreads();
  }
  if (tid == 0) {
    float offs = 0.5f * (s_red[0] + s_red2[0]);
    float rng  = 0.5f * (s_red[0] - s_red2[0]);
    s_b[3] = offs; s_b[4] = rng;
    offsetv[row] = offs;
    rangev [row] = rng;
    int tot = s_scan[255];
    ocount[row] = (tot < MAXO) ? tot : MAXO;
  }
  __syncthreads();
  const float offs = s_b[3], rng = s_b[4];
  const float inv  = (rng != 0.f) ? (1.f / rng) : 0.f;

  int pos = s_scan[tid] - cnt;   // exclusive prefix
  for (int i = 0; i < 16; ++i) {
    const int kk = base + i;
    float v  = s_orig[kk];
    float w0 = v - offs;
    bool  m  = (v <= lower) || (v >= upper);
    _Float16 q;
    if (m) {
      if (pos < MAXO) {
        oidx[(size_t)row * MAXO + pos] = kk;
        oval[(size_t)row * MAXO + pos] = w0;   // (weight - offset) at outlier
      }
      pos++;
      q = (_Float16)0.0f;                       // dense part is 0 -> nearest pole is nf[7]==0
    } else {
      float ws = w0 * inv;
      float bestd = 3.4e38f; int bi = 0;
      for (int p = 0; p < 16; ++p) {            // argmin keeps first minimum (ascending p)
        float d = fabsf(ws - s_nf[p]);
        if (d < bestd) { bestd = d; bi = p; }
      }
      q = (_Float16)s_nf[bi];                   // NF4 poles are exact in fp16
    }
    Qh[(size_t)row * ND + kk] = q;
  }
}

// ---------------------------------------------------------------------------
// Kernel 2: x -> fp16 hi/lo split (x = hi + lo, ~22 mantissa bits) + row sums
// ---------------------------------------------------------------------------
__global__ __launch_bounds__(256) void xprep_kernel(
    const float* __restrict__ X,
    _Float16* __restrict__ xhi, _Float16* __restrict__ xlo, float* __restrict__ rowsum)
{
  __shared__ float sr[256];
  const int m = blockIdx.x, tid = threadIdx.x;
  const float4* xr = (const float4*)(X + (size_t)m * ND);
  float s = 0.f;
  for (int i = tid; i < ND / 4; i += 256) {
    float4 v = xr[i];
    float vv[4] = {v.x, v.y, v.z, v.w};
    v4h hi, lo;
    for (int c = 0; c < 4; ++c) {
      _Float16 h = (_Float16)vv[c];
      hi[c] = h;
      lo[c] = (_Float16)(vv[c] - (float)h);
      s += vv[c];
    }
    ((v4h*)(xhi + (size_t)m * ND))[i] = hi;
    ((v4h*)(xlo + (size_t)m * ND))[i] = lo;
  }
  sr[tid] = s; __syncthreads();
  for (int st = 128; st > 0; st >>= 1) {
    if (tid < st) sr[tid] += sr[tid + st];
    __syncthreads();
  }
  if (tid == 0) rowsum[m] = sr[0];
}

// ---------------------------------------------------------------------------
// Kernel 3: WMMA GEMM  out[m,n] = range[n] * sum_k (xhi+xlo)[m,k]*Q[n,k]
//                               + offset[n] * rowsum[m]
// Block tile 128x128, BK=32, 8 waves (4 in M x 2 in N), wave tile 32x64.
// Double-buffered LDS filled by ASYNC global->LDS copies (ASYNCcnt).
// ---------------------------------------------------------------------------
static constexpr int BM  = 128;
static constexpr int BN  = 128;
static constexpr int BK  = 32;
static constexpr int LDK = 40;   // half stride -> 80B rows, 16B aligned frag loads

__global__ __launch_bounds__(256) void wmma_gemm_kernel(
    const _Float16* __restrict__ Ahi, const _Float16* __restrict__ Alo,
    const _Float16* __restrict__ Bq,
    const float* __restrict__ rangev, const float* __restrict__ offsetv,
    const float* __restrict__ rowsum, float* __restrict__ out)
{
  __shared__ __align__(16) _Float16 sAh[2][BM * LDK];
  __shared__ __align__(16) _Float16 sAl[2][BM * LDK];
  __shared__ __align__(16) _Float16 sB [2][BN * LDK];

  const int tid   = threadIdx.x;
  const int lane  = tid & 31;
  const int wave  = tid >> 5;
  const int waveM = wave & 3;    // 4 waves along M (32 rows each)
  const int waveN = wave >> 2;   // 2 waves along N (64 cols each)
  const int n0 = blockIdx.x * BN;
  const int m0 = blockIdx.y * BM;

  // tile loaders: thread -> one 32B half-row per matrix per K-step
  const int lr = tid >> 1;          // 0..127
  const int lc = (tid & 1) * 16;    // 0 | 16

  const uint64_t gA = (uint64_t)(uintptr_t)(Ahi + (size_t)(m0 + lr) * ND + lc);
  const uint64_t gL = (uint64_t)(uintptr_t)(Alo + (size_t)(m0 + lr) * ND + lc);
  const uint64_t gB = (uint64_t)(uintptr_t)(Bq  + (size_t)(n0 + lr) * ND + lc);

  uint32_t dA[2], dL[2], dB[2];
#pragma unroll
  for (int b = 0; b < 2; ++b) {
    dA[b] = lds_off_u32(&sAh[b][lr * LDK + lc]);
    dL[b] = lds_off_u32(&sAl[b][lr * LDK + lc]);
    dB[b] = lds_off_u32(&sB [b][lr * LDK + lc]);
  }

  // WMMA 16-bit operand layout: lanes 0-15 hold K in {0..7,16..23}, lanes 16-31 {8..15,24..31}
  const int rfrag = lane & 15;
  const int kbase = (lane >> 4) * 8;

  // prologue: DMA first K tile into buffer 0
  async_copy_b256(dA[0], gA);
  async_copy_b256(dL[0], gL);
  async_copy_b256(dB[0], gB);
  wait_async0();
  __syncthreads();

  v8f acc[2][4] = {};
  int cur = 0;
  for (int k0 = 0; k0 < ND; k0 += BK) {
    const int nxt = cur ^ 1;
    if (k0 + BK < ND) {            // overlap next tile's DMA with this tile's math
      const uint64_t koff = (uint64_t)(k0 + BK) * 2;  // bytes
      async_copy_b256(dA[nxt], gA + koff);
      async_copy_b256(dL[nxt], gL + koff);
      async_copy_b256(dB[nxt], gB + koff);
    }

    const _Float16* aB = &sAh[cur][0];
    const _Float16* lB = &sAl[cur][0];
    const _Float16* bBase = &sB[cur][0];

    v16h ah[2], al[2], bb[4];
#pragma unroll
    for (int i = 0; i < 2; ++i) {
      const _Float16* p = aB + (waveM * 32 + i * 16 + rfrag) * LDK + kbase;
      v8h x0 = *(const v8h*)p;
      v8h x1 = *(const v8h*)(p + 16);
      ah[i] = __builtin_shufflevector(x0, x1, 0,1,2,3,4,5,6,7,8,9,10,11,12,13,14,15);
      const _Float16* q = lB + (waveM * 32 + i * 16 + rfrag) * LDK + kbase;
      v8h y0 = *(const v8h*)q;
      v8h y1 = *(const v8h*)(q + 16);
      al[i] = __builtin_shufflevector(y0, y1, 0,1,2,3,4,5,6,7,8,9,10,11,12,13,14,15);
    }
#pragma unroll
    for (int j = 0; j < 4; ++j) {
      const _Float16* p = bBase + (waveN * 64 + j * 16 + rfrag) * LDK + kbase;
      v8h x0 = *(const v8h*)p;
      v8h x1 = *(const v8h*)(p + 16);
      bb[j] = __builtin_shufflevector(x0, x1, 0,1,2,3,4,5,6,7,8,9,10,11,12,13,14,15);
    }
#pragma unroll
    for (int i = 0; i < 2; ++i)
#pragma unroll
      for (int j = 0; j < 4; ++j) {
        acc[i][j] = __builtin_amdgcn_wmma_f32_16x16x32_f16(
            false, ah[i], false, bb[j], (short)0, acc[i][j], false, false);
        acc[i][j] = __builtin_amdgcn_wmma_f32_16x16x32_f16(
            false, al[i], false, bb[j], (short)0, acc[i][j], false, false);
      }

    wait_async0();        // my async copies into buf[nxt] have landed
    __syncthreads();      // everyone's copies landed + everyone done reading buf[cur]
    cur = nxt;
  }

  // Epilogue: C layout -> lanes 0-15: (M=r, N=lane); lanes 16-31: (M=8+r, N=lane-16)
  const int mhal = (lane >> 4) * 8;
#pragma unroll
  for (int i = 0; i < 2; ++i)
#pragma unroll
    for (int j = 0; j < 4; ++j) {
      const int nIdx = n0 + waveN * 64 + j * 16 + (lane & 15);
      const float rg = rangev[nIdx];
      const float of = offsetv[nIdx];
#pragma unroll
      for (int r = 0; r < 8; ++r) {
        const int mIdx = m0 + waveM * 32 + i * 16 + mhal + r;
        out[(size_t)mIdx * ND + nIdx] = acc[i][j][r] * rg + of * rowsum[mIdx];
      }
    }
}

// ---------------------------------------------------------------------------
// Kernel 4: exact fp32 sparse outlier correction: out[:,n] += x[:,k_j] * v_j
// grid = (n, m-chunk); deterministic (list order fixed, loop order fixed).
// ---------------------------------------------------------------------------
__global__ __launch_bounds__(256) void sparse_fix_kernel(
    const float* __restrict__ X, const int* __restrict__ ocount,
    const int* __restrict__ oidx, const float* __restrict__ oval,
    float* __restrict__ out)
{
  const int n = blockIdx.x;
  const int m = blockIdx.y * 256 + threadIdx.x;
  const int cnt = ocount[n];
  if (cnt == 0) return;
  float acc = 0.f;
  for (int j = 0; j < cnt; ++j) {
    acc += X[(size_t)m * ND + oidx[(size_t)n * MAXO + j]] * oval[(size_t)n * MAXO + j];
  }
  out[(size_t)m * ND + n] += acc;
}

// ---------------------------------------------------------------------------
extern "C" void kernel_launch(void* const* d_in, const int* in_sizes, int n_in,
                              void* d_out, int out_size, void* d_ws, size_t ws_size,
                              hipStream_t stream)
{
  (void)in_sizes; (void)n_in; (void)out_size; (void)ws_size;
  const float* X  = (const float*)d_in[0];   // [4096,4096] activations
  const float* W  = (const float*)d_in[1];   // [4096,4096] weight (row = out feature)
  const float* NF = (const float*)d_in[2];   // [16] NF4 poles
  float* out = (float*)d_out;

  char* ws = (char*)d_ws;
  size_t off = 0;
  auto alloc = [&](size_t bytes) -> void* {
    void* p = ws + off;
    off += (bytes + 255) & ~(size_t)255;
    return p;
  };
  _Float16* Qh      = (_Float16*)alloc((size_t)ND * ND * sizeof(_Float16)); // 32 MB
  _Float16* xhi     = (_Float16*)alloc((size_t)ND * ND * sizeof(_Float16)); // 32 MB
  _Float16* xlo     = (_Float16*)alloc((size_t)ND * ND * sizeof(_Float16)); // 32 MB
  float*    rangev  = (float*)alloc(ND * sizeof(float));
  float*    offsetv = (float*)alloc(ND * sizeof(float));
  float*    rowsum  = (float*)alloc(ND * sizeof(float));
  int*      ocount  = (int*)  alloc(ND * sizeof(int));
  int*      oidx    = (int*)  alloc((size_t)ND * MAXO * sizeof(int));
  float*    oval    = (float*)alloc((size_t)ND * MAXO * sizeof(float));

  nf4_quant_kernel<<<ND, 256, 0, stream>>>(W, NF, Qh, rangev, offsetv, ocount, oidx, oval);
  xprep_kernel<<<ND, 256, 0, stream>>>(X, xhi, xlo, rowsum);

  dim3 gg(ND / BN, ND / BM);
  wmma_gemm_kernel<<<gg, 256, 0, stream>>>(xhi, xlo, Qh, rangev, offsetv, rowsum, out);

  sparse_fix_kernel<<<dim3(ND, ND / 256), 256, 0, stream>>>(X, ocount, oidx, oval, out);
}
